// MultiheadAttention_27659589386393
// MI455X (gfx1250) — matrix operand order repaired and checked by reading, not compiled, not run on hardware
//
#include <hip/hip_runtime.h>

#define Bsz 2
#define Ssz 2048
#define Dsz 1024
#define Hsz 16
#define DKsz 64

typedef __attribute__((ext_vector_type(16))) __bf16 v16bf;
typedef __attribute__((ext_vector_type(16))) float  v16f;
typedef __attribute__((ext_vector_type(8)))  float  v8f;

// Load a 32-byte fragment as two contiguous 16-byte runs (global or LDS).
static __device__ __forceinline__ v16bf ld_frag_2x16B(const __bf16* p0,
                                                      const __bf16* p1) {
  union { uint4 q[2]; v16bf f; } u;
  u.q[0] = *(const uint4*)p0;
  u.q[1] = *(const uint4*)p1;
  return u.f;
}

// K-offset inside a 16-bit A-fragment (16x32) for vgpr-pair v (0..7), lane-half hi
static __device__ __forceinline__ int a_koff(int v, int hi) {
  return (v < 4 ? 2 * v : 16 + 2 * (v - 4)) + 8 * hi;
}

// Gather a B-fragment (32x16, 16-bit) from an f32 row-major matrix:
// b[2v]/b[2v+1] = M[k0 + hi*16 + 2v (+1)][col]  (stride = ldm floats)
static __device__ __forceinline__ v16bf ld_bfrag_f32_strided(const float* p,
                                                             int ldm) {
  v16f t;
#pragma unroll
  for (int vv = 0; vv < 16; ++vv) t[vv] = p[(size_t)vv * ldm];
  return __builtin_convertvector(t, v16bf);
}

// Load an A-fragment (16x32, 16-bit) from f32 memory: two runs of 8 floats.
static __device__ __forceinline__ v16bf ld_afrag_f32(const float* p, int hi) {
  union { float4 q[4]; v16f f; } u;
  const float* r0 = p + 16 * hi;
  const float* r1 = p + 16 * hi + 8;
  u.q[0] = *(const float4*)(r0);
  u.q[1] = *(const float4*)(r0 + 4);
  u.q[2] = *(const float4*)(r1);
  u.q[3] = *(const float4*)(r1 + 4);
  return __builtin_convertvector(u.f, v16bf);
}

// ---------------- Kernel 1: fused Q/K/V projections -------------------------
// x[4096,1024](f32) @ W[1024,1024](f32) -> bf16; Q,K stored [B,H,S,DK]
// (K pre-scaled by 1/sqrt(DK)); V stored transposed [B,H,DK,S].
__global__ __launch_bounds__(128)
void qkv_proj_kernel(const float* __restrict__ q, const float* __restrict__ k,
                     const float* __restrict__ v,
                     const float* __restrict__ Wq, const float* __restrict__ Wk,
                     const float* __restrict__ Wv,
                     __bf16* __restrict__ Qo, __bf16* __restrict__ Ko,
                     __bf16* __restrict__ Vo) {
  const int lane = threadIdx.x & 31;
  const int wid  = threadIdx.x >> 5;
  const int wglob = blockIdx.x * 4 + wid;   // 0..4095
  const int rst = wglob >> 6;               // row super-tile (64 rows) 0..63
  const int ct  = wglob & 63;               // col tile 0..63
  const int which = blockIdx.y;             // 0=Q,1=K,2=V
  const float* X = which == 0 ? q  : (which == 1 ? k  : v);
  const float* W = which == 0 ? Wq : (which == 1 ? Wk : Wv);
  __bf16*      O = which == 0 ? Qo : (which == 1 ? Ko : Vo);
  const float scale = (which == 1) ? 0.125f : 1.0f;

  const int m = lane & 15, hi = lane >> 4;
  const int row0 = rst * 64, col0 = ct * 16;

  v8f acc[4] = {{}, {}, {}, {}};
  for (int k0 = 0; k0 < Dsz; k0 += 32) {
    v16bf b = ld_bfrag_f32_strided(W + (size_t)(k0 + hi * 16) * Dsz + col0 + m,
                                   Dsz);
#pragma unroll
    for (int t = 0; t < 4; ++t) {
      v16bf a = ld_afrag_f32(X + (size_t)(row0 + t * 16 + m) * Dsz + k0, hi);
      acc[t] = __builtin_amdgcn_wmma_f32_16x16x32_bf16(false, a, false, b,
                                                       (short)0, acc[t],
                                                       false, false);
    }
  }
  const int col = col0 + m;
  const int h = col / DKsz, d = col % DKsz;
#pragma unroll
  for (int t = 0; t < 4; ++t) {
#pragma unroll
    for (int r = 0; r < 8; ++r) {
      int row = row0 + t * 16 + r + 8 * hi;
      int bb = row / Ssz, s = row % Ssz;
      size_t idx;
      if (which == 2)  // V transposed: [B,H,DK,S]
        idx = (((size_t)(bb * Hsz + h) * DKsz) + d) * Ssz + s;
      else             // Q,K: [B,H,S,DK]
        idx = (((size_t)(bb * Hsz + h) * Ssz) + s) * DKsz + d;
      O[idx] = (__bf16)(acc[t][r] * scale);
    }
  }
}

// ---------------- Kernel 2: causal flash attention --------------------------
#define PSTRIDE 40  // bf16 elements per LDS row (conflict-free, 16B-aligned runs)
__global__ __launch_bounds__(128)
void flash_attn_kernel(const __bf16* __restrict__ Q, const __bf16* __restrict__ K,
                       const __bf16* __restrict__ Vt, __bf16* __restrict__ AO) {
  __shared__ __bf16 plds[4][16 * PSTRIDE];
  const int lane = threadIdx.x & 31;
  const int wid  = threadIdx.x >> 5;
  const int n = lane & 15, hi = lane >> 4;
  const int qt = blockIdx.x * 4 + wid;   // 0..127
  const int bh = blockIdx.y;             // 0..31
  const int q0 = qt * 16;
  const __bf16* Qp = Q  + (size_t)bh * Ssz * DKsz;
  const __bf16* Kp = K  + (size_t)bh * Ssz * DKsz;
  const __bf16* Vp = Vt + (size_t)bh * DKsz * Ssz;  // [DK][S]

  // Q fragments: 16x64 = two 16x32 A-fragments (contiguous per lane)
  const __bf16* qp = Qp + (size_t)(q0 + n) * DKsz;
  v16bf qa0 = ld_frag_2x16B(qp + 16 * hi, qp + 16 * hi + 8);
  v16bf qa1 = ld_frag_2x16B(qp + 32 + 16 * hi, qp + 40 + 16 * hi);

  v8f o0 = {}, o1 = {}, o2 = {}, o3 = {};
  float mrow[8], lrow[8];
#pragma unroll
  for (int r = 0; r < 8; ++r) { mrow[r] = -3.0e38f; lrow[r] = 0.f; }

  for (int j0 = 0; j0 <= q0 + 15; j0 += 32) {
    // prefetch next KV block while we work on this one
    __builtin_prefetch(Kp + (size_t)(j0 + 32 + n) * DKsz, 0, 0);
    __builtin_prefetch(Vp + (size_t)(hi * 16 + n) * Ssz + j0 + 32, 0, 0);

    // ---- S = Q K^T for two 16-key sub-tiles ----
    v8f sc[2];
#pragma unroll
    for (int sub = 0; sub < 2; ++sub) {
      const __bf16* kp = Kp + (size_t)(j0 + sub * 16 + n) * DKsz + hi * 16;
      v16bf kb0 = ld_frag_2x16B(kp, kp + 8);            // dk 0..31 half
      v16bf kb1 = ld_frag_2x16B(kp + 32, kp + 40);      // dk 32..63 half
      v8f t = {};
      t = __builtin_amdgcn_wmma_f32_16x16x32_bf16(false, qa0, false, kb0,
                                                  (short)0, t, false, false);
      t = __builtin_amdgcn_wmma_f32_16x16x32_bf16(false, qa1, false, kb1,
                                                  (short)0, t, false, false);
      sc[sub] = t;
    }
    // ---- causal mask + online softmax (row r+8*hi spans 16 lanes) ----
    float alpha[8];
#pragma unroll
    for (int r = 0; r < 8; ++r) {
      int qrow = q0 + r + 8 * hi;
      float x0 = (j0 + n      <= qrow) ? sc[0][r] : -1e9f;
      float x1 = (j0 + 16 + n <= qrow) ? sc[1][r] : -1e9f;
      float t = fmaxf(x0, x1);
      t = fmaxf(t, __shfl_xor(t, 1));
      t = fmaxf(t, __shfl_xor(t, 2));
      t = fmaxf(t, __shfl_xor(t, 4));
      t = fmaxf(t, __shfl_xor(t, 8));
      float mn = fmaxf(mrow[r], t);
      alpha[r] = __expf(mrow[r] - mn);
      mrow[r] = mn;
      float p0 = __expf(x0 - mn), p1 = __expf(x1 - mn);
      float rs = p0 + p1;
      rs += __shfl_xor(rs, 1);
      rs += __shfl_xor(rs, 2);
      rs += __shfl_xor(rs, 4);
      rs += __shfl_xor(rs, 8);
      lrow[r] = lrow[r] * alpha[r] + rs;
      plds[wid][(r + 8 * hi) * PSTRIDE + n]      = (__bf16)p0;
      plds[wid][(r + 8 * hi) * PSTRIDE + 16 + n] = (__bf16)p1;
    }
#pragma unroll
    for (int r = 0; r < 8; ++r) {
      o0[r] *= alpha[r]; o1[r] *= alpha[r];
      o2[r] *= alpha[r]; o3[r] *= alpha[r];
    }
    // ---- P fragment (16x32) from wave-private LDS (DS ops in-order) ----
    const __bf16* pr = &plds[wid][n * PSTRIDE];
    v16bf pa = ld_frag_2x16B(pr + 16 * hi, pr + 16 * hi + 8);
    // ---- O += P V over four 16-wide d tiles (V transposed: contiguous) ----
#pragma unroll
    for (int nt = 0; nt < 4; ++nt) {
      const __bf16* vp = Vp + (size_t)(nt * 16 + n) * Ssz + j0 + hi * 16;
      v16bf vb = ld_frag_2x16B(vp, vp + 8);
      v8f* op = nt == 0 ? &o0 : nt == 1 ? &o1 : nt == 2 ? &o2 : &o3;
      *op = __builtin_amdgcn_wmma_f32_16x16x32_bf16(false, pa, false, vb,
                                                    (short)0, *op, false, false);
    }
  }
  // ---- epilogue: O/l -> bf16 attn-out stored [B,S,H*DK] ----
  const int bb = bh / Hsz, h = bh % Hsz;
#pragma unroll
  for (int r = 0; r < 8; ++r) {
    int row = q0 + r + 8 * hi;
    float inv = 1.0f / lrow[r];
    size_t base = ((size_t)(bb * Ssz + row)) * Dsz + h * DKsz + n;
    AO[base +  0] = (__bf16)(o0[r] * inv);
    AO[base + 16] = (__bf16)(o1[r] * inv);
    AO[base + 32] = (__bf16)(o2[r] * inv);
    AO[base + 48] = (__bf16)(o3[r] * inv);
  }
}

// ---------------- Kernel 3: output projection -------------------------------
__global__ __launch_bounds__(128)
void out_proj_kernel(const __bf16* __restrict__ A, const float* __restrict__ Wo,
                     float* __restrict__ out) {
  const int lane = threadIdx.x & 31;
  const int wid  = threadIdx.x >> 5;
  const int wglob = blockIdx.x * 4 + wid;   // 0..4095
  const int rst = wglob >> 6, ct = wglob & 63;
  const int m = lane & 15, hi = lane >> 4;
  const int row0 = rst * 64, col0 = ct * 16;
  v8f acc[4] = {{}, {}, {}, {}};
  for (int k0 = 0; k0 < Dsz; k0 += 32) {
    v16bf b = ld_bfrag_f32_strided(Wo + (size_t)(k0 + hi * 16) * Dsz + col0 + m,
                                   Dsz);
#pragma unroll
    for (int t = 0; t < 4; ++t) {
      const __bf16* ap = A + (size_t)(row0 + t * 16 + m) * Dsz + k0;
      v16bf a = ld_frag_2x16B(ap + 16 * hi, ap + 16 * hi + 8);
      acc[t] = __builtin_amdgcn_wmma_f32_16x16x32_bf16(false, a, false, b,
                                                       (short)0, acc[t],
                                                       false, false);
    }
  }
#pragma unroll
  for (int t = 0; t < 4; ++t) {
#pragma unroll
    for (int r = 0; r < 8; ++r) {
      out[(size_t)(row0 + t * 16 + r + 8 * hi) * Dsz + col0 + m] = acc[t][r];
    }
  }
}

extern "C" void kernel_launch(void* const* d_in, const int* in_sizes, int n_in,
                              void* d_out, int out_size, void* d_ws, size_t ws_size,
                              hipStream_t stream) {
  const float* q  = (const float*)d_in[0];
  const float* k  = (const float*)d_in[1];
  const float* v  = (const float*)d_in[2];
  const float* Wq = (const float*)d_in[3];
  const float* Wk = (const float*)d_in[4];
  const float* Wv = (const float*)d_in[5];
  const float* Wo = (const float*)d_in[6];
  // d_in[7] = causal mask, implemented analytically in-kernel
  float* out = (float*)d_out;

  const size_t nQKV = (size_t)Bsz * Hsz * Ssz * DKsz;  // 4M elements
  __bf16* Qbf = (__bf16*)d_ws;
  __bf16* Kbf = Qbf + nQKV;
  __bf16* Vtb = Kbf + nQKV;  // transposed [B,H,DK,S]
  __bf16* Abf = Vtb + nQKV;  // attention output [B*S, D]

  dim3 gp(1024, 3);  // 4096 waves/projection = 64 x 64 (64-row x 16-col tiles)
  qkv_proj_kernel<<<gp, 128, 0, stream>>>(q, k, v, Wq, Wk, Wv, Qbf, Kbf, Vtb);

  dim3 ga(32, 32);   // (S/16/4 qtile-groups) x (B*H)
  flash_attn_kernel<<<ga, 128, 0, stream>>>(Qbf, Kbf, Vtb, Abf);

  out_proj_kernel<<<1024, 128, 0, stream>>>(Abf, Wo, out);
}
